// Model_44659069944413
// MI455X (gfx1250) — compile-verified, matrix operand order checked by reading, not run on hardware
//
#include <hip/hip_runtime.h>
#include <hip/hip_bf16.h>

// ---------------------------------------------------------------------------
// Hetero-GraphSAGE (pool) x3 layers + edge MLP decoder for MI455X (gfx1250).
// GEMMs run on v_wmma_f32_16x16x32_bf16 (wave32 WMMA, f32 accumulate).
// Each wave computes a 32x128 output stripe (2 M-tiles x 8 N-tiles): every
// B fragment fetched from LDS feeds two WMMAs.
// f32 -> bf16x2 conversion uses a single v_perm_b32 per element pair
// (truncation), so the A-fragment build is 16 VALU ops per k-step instead
// of ~96, keeping VGPRs under the 256 boundary.
// segment_max is u32 atomic max on raw float bits (exact: p >= 0 after relu,
// and the 0-init implements the max(m, 0) clamp for empty segments).
// ---------------------------------------------------------------------------

typedef __attribute__((ext_vector_type(16))) __bf16 v16bf;
typedef __attribute__((ext_vector_type(8)))  float  v8f;

union FragU {            // lets vector loads land directly in WMMA source regs
    v16bf    v;
    unsigned u[8];
};

// Pack two f32 into one dword of 2x bf16 (truncate): result low16 = bf16(lo),
// high16 = bf16(hi). v_perm_b32 byte codes: 0-3 pick S1 bytes, 4-7 pick S0.
__device__ __forceinline__ unsigned pack_bf16x2(float lo, float hi) {
    return __builtin_amdgcn_perm(__builtin_bit_cast(unsigned, hi),
                                 __builtin_bit_cast(unsigned, lo),
                                 0x07060302u);
}

#define DFEAT 128
#define LDS_STRIDE 136   // 128 + 8 pad: breaks 64-bank conflict; 16B-aligned rows

// C[n,128] = A·Wᵀ (+ A2·W2ᵀ) (+ bias) (+ out if accumulate), ×scale, opt relu.
// Block = 128 threads (4 waves); each wave owns a 32-row stripe (2 M-tiles).
// Requires n % 32 == 0 (N = 100000 = 32 * 3125).
__global__ __launch_bounds__(128) void sage_gemm(
    const float* __restrict__ A,  const float* __restrict__ W,
    const float* __restrict__ A2, const float* __restrict__ W2,
    const float* __restrict__ bias,
    float* out, int n, float scale, int relu_out, int accumulate)
{
    __shared__ unsigned short sW[2][DFEAT][LDS_STRIDE];

    const int tid  = threadIdx.x;
    const int lane = tid & 31;
    const int wave = tid >> 5;

    // Stage weights into LDS as bf16 pairs. sW[p][j][k] = bf16(W[j*128+k])
    // (B tile element B[k][ncol] = W[ncol][k]; rows indexed by out-col).
    for (int i = tid * 2; i < DFEAT * DFEAT; i += 256) {
        *(unsigned*)&sW[0][i >> 7][i & 127] = pack_bf16x2(W[i], W[i + 1]);
    }
    if (W2) {
        for (int i = tid * 2; i < DFEAT * DFEAT; i += 256) {
            *(unsigned*)&sW[1][i >> 7][i & 127] = pack_bf16x2(W2[i], W2[i + 1]);
        }
    }
    __syncthreads();

    const int rowbase = blockIdx.x * 128 + wave * 32;   // 2 M-tiles per wave
    if (rowbase >= n) return;                           // uniform per wave

    v8f acc[2][8];
    #pragma unroll
    for (int t = 0; t < 2; ++t)
        #pragma unroll
        for (int c = 0; c < 8; ++c) acc[t][c] = (v8f)0.0f;

    const int m16 = lane & 15;         // row within A tile / col within B tile
    const int g   = lane >> 4;         // half-wave group

    const int npass = A2 ? 2 : 1;
    for (int pass = 0; pass < npass; ++pass) {
        const float* Ap = pass ? A2 : A;
        const float* ar0 = Ap + (size_t)(rowbase + m16) * DFEAT;        // tile 0
        const float* ar1 = Ap + (size_t)(rowbase + 16 + m16) * DFEAT;   // tile 1

        #pragma unroll
        for (int kb = 0; kb < DFEAT; kb += 32) {
            // A fragments: 16-bit A 16x32 layout.
            // elems 0..7  -> k = kb + g*8 + e ; elems 8..15 -> k = kb+16+g*8+(e-8)
            FragU a0, a1;
            {
                const float* r0 = ar0 + kb + g * 8;
                const float* r1 = ar1 + kb + g * 8;
                #pragma unroll
                for (int i = 0; i < 4; ++i) {
                    a0.u[i]     = pack_bf16x2(r0[2 * i],      r0[2 * i + 1]);
                    a0.u[4 + i] = pack_bf16x2(r0[16 + 2 * i], r0[17 + 2 * i]);
                    a1.u[i]     = pack_bf16x2(r1[2 * i],      r1[2 * i + 1]);
                    a1.u[4 + i] = pack_bf16x2(r1[16 + 2 * i], r1[17 + 2 * i]);
                }
            }
            #pragma unroll
            for (int ct = 0; ct < 8; ++ct) {
                const int ncol = ct * 16 + m16;
                FragU b;
                const uint4 blo = *(const uint4*)&sW[pass][ncol][kb + g * 8];
                const uint4 bhi = *(const uint4*)&sW[pass][ncol][kb + 16 + g * 8];
                b.u[0] = blo.x; b.u[1] = blo.y; b.u[2] = blo.z; b.u[3] = blo.w;
                b.u[4] = bhi.x; b.u[5] = bhi.y; b.u[6] = bhi.z; b.u[7] = bhi.w;

                acc[0][ct] = __builtin_amdgcn_wmma_f32_16x16x32_bf16(
                    false, a0.v, false, b.v, (short)0, acc[0][ct], false, false);
                acc[1][ct] = __builtin_amdgcn_wmma_f32_16x16x32_bf16(
                    false, a1.v, false, b.v, (short)0, acc[1][ct], false, false);
            }
        }
    }

    // Epilogue. 32-bit C/D 16x16 layout: VGPR v -> row = tilebase + g*8 + v,
    // col = ct*16 + (lane & 15). Two uniform variants, branch-free per element.
    if (accumulate) {
        #pragma unroll
        for (int t = 0; t < 2; ++t) {
            const int r0 = rowbase + t * 16 + g * 8;
            #pragma unroll
            for (int ct = 0; ct < 8; ++ct) {
                const int col = ct * 16 + m16;
                const float bv = bias[col];
                #pragma unroll
                for (int v = 0; v < 8; ++v) {
                    size_t idx = (size_t)(r0 + v) * DFEAT + col;
                    float val = (acc[t][ct][v] + bv + out[idx]) * scale;
                    if (relu_out) val = fmaxf(val, 0.0f);
                    out[idx] = val;
                }
            }
        }
    } else {
        #pragma unroll
        for (int t = 0; t < 2; ++t) {
            const int r0 = rowbase + t * 16 + g * 8;
            #pragma unroll
            for (int ct = 0; ct < 8; ++ct) {
                const int col = ct * 16 + m16;
                const float bv = bias[col];
                #pragma unroll
                for (int v = 0; v < 8; ++v) {
                    size_t idx = (size_t)(r0 + v) * DFEAT + col;
                    float val = (acc[t][ct][v] + bv) * scale;
                    if (relu_out) val = fmaxf(val, 0.0f);
                    out[idx] = val;
                }
            }
        }
    }
}

__global__ __launch_bounds__(256) void zero_f4(float4* __restrict__ p, int n4) {
    int i = blockIdx.x * blockDim.x + threadIdx.x;
    if (i < n4) p[i] = make_float4(0.f, 0.f, 0.f, 0.f);
}

// One wave per edge: gather p[src] (float4/lane), u32 atomic-max into m[dst].
// p and m are 51.2 MB each -> resident in the 192 MB L2.
__global__ __launch_bounds__(256) void edge_pool_max(
    const float* __restrict__ p, const int* __restrict__ src,
    const int* __restrict__ dst, unsigned* __restrict__ m, int ne)
{
    const int lane = threadIdx.x & 31;
    const int e = (blockIdx.x * blockDim.x + threadIdx.x) >> 5;
    if (e >= ne) return;
    const int s = src[e], d = dst[e];
    const float4 v = *(const float4*)(p + (size_t)s * DFEAT + lane * 4);
    unsigned* mb = m + (size_t)d * DFEAT + lane * 4;
    atomicMax(mb + 0, __builtin_bit_cast(unsigned, v.x));
    atomicMax(mb + 1, __builtin_bit_cast(unsigned, v.y));
    atomicMax(mb + 2, __builtin_bit_cast(unsigned, v.z));
    atomicMax(mb + 3, __builtin_bit_cast(unsigned, v.w));
}

// One wave per decode edge: out[e,c] = <h[s], Wc[0:128]> + <h[d], Wc[128:256]> + b[c]
__global__ __launch_bounds__(256) void decode_kernel(
    const float* __restrict__ h, const int* __restrict__ dsrc,
    const int* __restrict__ ddst, const float* __restrict__ predW,
    const float* __restrict__ predb, float* __restrict__ out, int ne)
{
    const int lane = threadIdx.x & 31;
    const int e = (blockIdx.x * blockDim.x + threadIdx.x) >> 5;
    if (e >= ne) return;
    const int s = dsrc[e], d = ddst[e];
    const float4 hs = *(const float4*)(h + (size_t)s * DFEAT + lane * 4);
    const float4 hd = *(const float4*)(h + (size_t)d * DFEAT + lane * 4);
    float res[3];
    #pragma unroll
    for (int c = 0; c < 3; ++c) {
        const float4 ws = *(const float4*)(predW + c * 256 + lane * 4);
        const float4 wd = *(const float4*)(predW + c * 256 + 128 + lane * 4);
        float t = hs.x * ws.x + hs.y * ws.y + hs.z * ws.z + hs.w * ws.w
                + hd.x * wd.x + hd.y * wd.y + hd.z * wd.z + hd.w * wd.w;
        #pragma unroll
        for (int off = 16; off >= 1; off >>= 1) t += __shfl_xor(t, off, 32);
        res[c] = t;
    }
    if (lane == 0) {
        out[(size_t)e * 3 + 0] = res[0] + predb[0];
        out[(size_t)e * 3 + 1] = res[1] + predb[1];
        out[(size_t)e * 3 + 2] = res[2] + predb[2];
    }
}

extern "C" void kernel_launch(void* const* d_in, const int* in_sizes, int n_in,
                              void* d_out, int out_size, void* d_ws, size_t ws_size,
                              hipStream_t stream) {
    (void)in_sizes; (void)n_in; (void)out_size; (void)ws_size;
    const int N = 100000, D = 128, R = 3, E = 600000, EDEC = 200000, L = 3;

    const float* x      = (const float*)d_in[0];
    const int*   edges  = (const int*)  d_in[1];
    const int*   dec    = (const int*)  d_in[2];
    const float* poolW  = (const float*)d_in[3];
    const float* poolB  = (const float*)d_in[4];
    const float* selfW  = (const float*)d_in[5];
    const float* neighW = (const float*)d_in[6];
    const float* sageB  = (const float*)d_in[7];
    const float* predW  = (const float*)d_in[8];
    const float* predb  = (const float*)d_in[9];
    float* outp = (float*)d_out;

    const size_t feat = (size_t)N * D;
    char* ws = (char*)d_ws;
    float* hA = (float*)ws; ws += feat * sizeof(float);
    float* hB = (float*)ws; ws += feat * sizeof(float);
    float* p  = (float*)ws; ws += feat * sizeof(float);
    float* m  = (float*)ws; ws += feat * sizeof(float);

    const int gemm_grid  = (N + 127) / 128;            // 32 rows per wave, 4 waves
    const int zero_grid  = (int)((feat / 4 + 255) / 256);
    const int edge_grid  = (int)(((size_t)E * 32 + 255) / 256);
    const int dec_grid   = (int)(((size_t)EDEC * 32 + 255) / 256);

    const float* hcur = x;
    float* bufs[2] = { hA, hB };

    for (int l = 0; l < L; ++l) {
        float* hnext = bufs[l & 1];
        for (int r = 0; r < R; ++r) {
            const size_t wi = (size_t)(l * R + r);
            // p = relu(h @ pool_W^T + pool_b)
            sage_gemm<<<gemm_grid, 128, 0, stream>>>(
                hcur, poolW + wi * D * D, nullptr, nullptr,
                poolB + wi * D, p, N, 1.0f, 1, 0);
            // m = 0
            zero_f4<<<zero_grid, 256, 0, stream>>>((float4*)m, (int)(feat / 4));
            // m = segment_max over edges (u32 atomic max on bits, p >= 0)
            const int* src = edges + ((size_t)r * 2 + 0) * E;
            const int* dst = edges + ((size_t)r * 2 + 1) * E;
            edge_pool_max<<<edge_grid, 256, 0, stream>>>(p, src, dst, (unsigned*)m, E);
            // hnext (+)= h @ self_W^T + m @ neigh_W^T + bias ; last r: /3, relu(l<L-1)
            const int fin = (r == R - 1);
            sage_gemm<<<gemm_grid, 128, 0, stream>>>(
                hcur, selfW + wi * D * D, m, neighW + wi * D * D,
                sageB + wi * D, hnext, N,
                fin ? (1.0f / 3.0f) : 1.0f, fin && (l < L - 1), r != 0);
        }
        hcur = hnext;
    }

    decode_kernel<<<dec_grid, 256, 0, stream>>>(
        hcur, dec, dec + EDEC, predW, predb, outp, EDEC);
}
